// GraphTransformer_82257213653583
// MI455X (gfx1250) — matrix operand order repaired and checked by reading, not compiled, not run on hardware
//
#include <hip/hip_runtime.h>
#include <hip/hip_fp16.h>

typedef __attribute__((ext_vector_type(16))) _Float16 v16h;
typedef __attribute__((ext_vector_type(8)))  float    v8f;

#define NNODES 50000
#define NEDGES 800000
#define FIN    64
#define HID    32
#define NH     4
#define DCAT   128   // HID*NH
#define NG     1000

#define NTILES 3125              // NNODES / 16, exact
#define WAVES_PER_BLK 8
#define GEMM_GRID ((NTILES + WAVES_PER_BLK - 1) / WAVES_PER_BLK)  // 391

// Async global->LDS path (gfx1250 GLOBAL_LOAD_ASYNC_TO_LDS_B128), if exposed.
#if __has_builtin(__builtin_amdgcn_global_load_async_to_lds_b128) && \
    __has_builtin(__builtin_amdgcn_s_wait_asynccnt)
#define USE_ASYNC_LDS 1
typedef int v4i __attribute__((ext_vector_type(4)));
typedef __attribute__((address_space(1))) v4i as1_v4i;   // global v4i
typedef __attribute__((address_space(3))) v4i as3_v4i;   // LDS v4i
#else
#define USE_ASYNC_LDS 0
#endif

// ---------------------------------------------------------------------------
// Weight convert: Wt[m*K + k] = (f16) W[k*M + m]   (transpose + f32->f16)
// ---------------------------------------------------------------------------
__global__ void convert_w(const float* __restrict__ W, _Float16* __restrict__ Wt,
                          int K, int M) {
    int idx = blockIdx.x * blockDim.x + threadIdx.x;
    if (idx >= K * M) return;
    int k = idx / M, m = idx % M;
    Wt[(size_t)m * K + k] = (_Float16)W[idx];
}

// ---------------------------------------------------------------------------
// GEMM: Y[N,M] = X[N,K] @ W[K,M] + bias   via v_wmma_f32_16x16x32_f16
// 256-thread blocks (8 waves).  The whole f16 weight matrix (<=32KB) is staged
// into LDS once per block (async-to-LDS when available), then each wave
// computes one 16-row stripe across all M/16 col-tiles, reading B fragments
// from LDS (row stride padded by 8 halves -> conflict-free per half-wave).
//
// A-frag lane layout (16-bit A 16x32, ISA 7.12.2):
//   lanes 0-15 : row=lane,    K = {0..7, 16..23}
//   lanes 16-31: row=lane-16, K = {8..15, 24..31}
// B-frag as loaded here, same k as A load: VGPR v, lanes 0-15: K=2v,2v+1 ; lanes 16-31: K=16+2v,+17; N=lane&15
// C/D: VGPR r -> row r + 8*(lane>=16), col = lane&15
// ---------------------------------------------------------------------------
template <int MT, int KT>
__device__ __forceinline__ void gemm_lds_body(const float* __restrict__ X,
                                              const _Float16* __restrict__ Wt,
                                              const float* __restrict__ bias,
                                              float* __restrict__ Y,
                                              _Float16* __restrict__ WS) {
    constexpr int M    = MT * 16;
    constexpr int LROW = KT + 8;          // padded LDS row stride (halves)
    const int tid  = threadIdx.x;
    const int wave = tid >> 5;
    const int lane = tid & 31;
    const int hi   = lane >> 4;           // 0 or 1 (half-wave)
    const int lr   = lane & 15;

    // ---- stage full weight matrix [M, KT] (f16) into LDS, 16B per transfer
    constexpr int GROUPS = M * (KT / 8);  // exact multiple of 256 for all inst.
#pragma unroll
    for (int g = tid; g < GROUPS; g += 256) {
        int row = g / (KT / 8);
        int ko  = (g % (KT / 8)) * 8;
        const _Float16* gp = Wt + (size_t)row * KT + ko;
        _Float16*       lp = WS + row * LROW + ko;
#if USE_ASYNC_LDS
        __builtin_amdgcn_global_load_async_to_lds_b128(
            (as1_v4i*)gp, (as3_v4i*)lp, 0, 0);
#else
        *(uint4*)lp = *(const uint4*)gp;
#endif
    }
#if USE_ASYNC_LDS
    __builtin_amdgcn_s_wait_asynccnt(0);
#endif
    __syncthreads();

    const int tile = blockIdx.x * WAVES_PER_BLK + wave;   // wave-uniform
    if (tile >= NTILES) return;
    const int m0 = tile * 16;

    const float* xrow = X + (size_t)(m0 + lr) * KT;

    v8f c[MT];
#pragma unroll
    for (int t = 0; t < MT; ++t)
#pragma unroll
        for (int r = 0; r < 8; ++r) c[t][r] = 0.0f;

    for (int kb = 0; kb < KT; kb += 32) {
        if (kb + 32 < KT) __builtin_prefetch(xrow + kb + 32, 0, 3);
        // ---- A fragment: 16 contiguous f32 (K = 16*hi .. 16*hi+15) -> f16
        float4 f0 = *(const float4*)(xrow + kb + hi * 16);
        float4 f1 = *(const float4*)(xrow + kb + hi * 16 + 4);
        float4 f2 = *(const float4*)(xrow + kb + hi * 16 + 8);
        float4 f3 = *(const float4*)(xrow + kb + hi * 16 + 12);
        v16h a;
        a[0]  = (_Float16)f0.x; a[1]  = (_Float16)f0.y;
        a[2]  = (_Float16)f0.z; a[3]  = (_Float16)f0.w;
        a[4]  = (_Float16)f1.x; a[5]  = (_Float16)f1.y;
        a[6]  = (_Float16)f1.z; a[7]  = (_Float16)f1.w;
        a[8]  = (_Float16)f2.x; a[9]  = (_Float16)f2.y;
        a[10] = (_Float16)f2.z; a[11] = (_Float16)f2.w;
        a[12] = (_Float16)f3.x; a[13] = (_Float16)f3.y;
        a[14] = (_Float16)f3.z; a[15] = (_Float16)f3.w;

#pragma unroll
        for (int t = 0; t < MT; ++t) {
            // B fragment: 16 contiguous halves from LDS (ds_load_b128 x2)
            const _Float16* wl = WS + (t * 16 + lr) * LROW + kb + hi * 16;
            v16h b = *(const v16h*)wl;
            c[t] = __builtin_amdgcn_wmma_f32_16x16x32_f16(
                false, a, false, b, (short)0, c[t], false, false);
        }
    }

#pragma unroll
    for (int t = 0; t < MT; ++t) {
        float bn = bias[t * 16 + lr];
#pragma unroll
        for (int r = 0; r < 8; ++r) {
            int m = m0 + r + hi * 8;
            Y[(size_t)m * M + t * 16 + lr] = c[t][r] + bn;
        }
    }
}

__global__ void __launch_bounds__(256)
gemm_m128_k64(const float* X, const _Float16* Wt, const float* bias, float* Y) {
    __shared__ _Float16 WS[128 * (64 + 8)];
    gemm_lds_body<8, 64>(X, Wt, bias, Y, WS);
}
__global__ void __launch_bounds__(256)
gemm_m128_k128(const float* X, const _Float16* Wt, const float* bias, float* Y) {
    __shared__ _Float16 WS[128 * (128 + 8)];
    gemm_lds_body<8, 128>(X, Wt, bias, Y, WS);
}
__global__ void __launch_bounds__(256)
gemm_m32_k128(const float* X, const _Float16* Wt, const float* bias, float* Y) {
    __shared__ _Float16 WS[32 * (128 + 8)];
    gemm_lds_body<2, 128>(X, Wt, bias, Y, WS);
}

// ---------------------------------------------------------------------------
// Order-preserving float <-> uint key (for atomic segment-max)
// ---------------------------------------------------------------------------
__device__ __forceinline__ unsigned f2key(float f) {
    unsigned u = __float_as_uint(f);
    return (u & 0x80000000u) ? ~u : (u | 0x80000000u);
}
__device__ __forceinline__ float key2f(unsigned k) {
    unsigned u = (k & 0x80000000u) ? (k & 0x7fffffffu) : ~k;
    return __uint_as_float(u);
}

// ---------------------------------------------------------------------------
// Pass A: per-edge logits + segment max over dst
// ---------------------------------------------------------------------------
__global__ void edge_logits_max(const float* __restrict__ q,
                                const float* __restrict__ k,
                                const int* __restrict__ src,
                                const int* __restrict__ dst,
                                float* __restrict__ P,
                                unsigned* __restrict__ Mkey) {
    int e = blockIdx.x * blockDim.x + threadIdx.x;
    if (e >= NEDGES) return;
    int s = src[e], d = dst[e];
    const float4* qe = (const float4*)(q + (size_t)d * DCAT);
    const float4* ke = (const float4*)(k + (size_t)s * DCAT);
#pragma unroll
    for (int h = 0; h < NH; ++h) {
        float acc = 0.0f;
#pragma unroll
        for (int t = 0; t < 8; ++t) {
            float4 a = qe[h * 8 + t], b = ke[h * 8 + t];
            acc += a.x * b.x + a.y * b.y + a.z * b.z + a.w * b.w;
        }
        float lg = acc * 0.17677669529663687f;  // 1/sqrt(32)
        P[(size_t)e * NH + h] = lg;
        atomicMax(&Mkey[(size_t)d * NH + h], f2key(lg));
    }
}

// ---------------------------------------------------------------------------
// Pass B: p = exp(logit - m[dst]); Z[dst] += p      (one thread per edge*head)
// ---------------------------------------------------------------------------
__global__ void edge_softmax_num(const int* __restrict__ dst,
                                 const unsigned* __restrict__ Mkey,
                                 float* __restrict__ P,
                                 float* __restrict__ Z) {
    int i = blockIdx.x * blockDim.x + threadIdx.x;
    if (i >= NEDGES * NH) return;
    int e = i >> 2, h = i & 3;
    int d = dst[e];
    float p = __expf(P[i] - key2f(Mkey[(size_t)d * NH + h]));
    P[i] = p;
    atomicAdd(&Z[(size_t)d * NH + h], p);
}

// ---------------------------------------------------------------------------
// Pass C: ATT[dst,h,:] += (p/Z[dst,h]) * V[src,h,:]  (one thread per edge*head)
// ---------------------------------------------------------------------------
__global__ void edge_aggregate(const int* __restrict__ src,
                               const int* __restrict__ dst,
                               const float* __restrict__ P,
                               const float* __restrict__ Z,
                               const float* __restrict__ V,
                               float* __restrict__ ATT) {
    int i = blockIdx.x * blockDim.x + threadIdx.x;
    if (i >= NEDGES * NH) return;
    int e = i >> 2, h = i & 3;
    int s = src[e], d = dst[e];
    float coef = P[i] / Z[(size_t)d * NH + h];
    const float* vrow = V + (size_t)s * DCAT + h * HID;
    float* arow = ATT + (size_t)d * DCAT + h * HID;
#pragma unroll
    for (int j = 0; j < HID; ++j) atomicAdd(&arow[j], coef * vrow[j]);
}

// ---------------------------------------------------------------------------
// Combine: concat layers (y = att + skip, relu) / final layer (head mean)
// ---------------------------------------------------------------------------
__global__ void combine_concat_relu(const float* __restrict__ ATT,
                                    const float* __restrict__ S,
                                    float* __restrict__ Y) {
    int i = blockIdx.x * blockDim.x + threadIdx.x;
    if (i >= NNODES * DCAT) return;
    Y[i] = fmaxf(ATT[i] + S[i], 0.0f);
}

__global__ void combine_mean(const float* __restrict__ ATT,
                             const float* __restrict__ S,
                             float* __restrict__ Y) {
    int i = blockIdx.x * blockDim.x + threadIdx.x;
    if (i >= NNODES * HID) return;
    int n = i / HID, j = i % HID;
    const float* a = ATT + (size_t)n * DCAT;
    Y[i] = 0.25f * (a[j] + a[HID + j] + a[2 * HID + j] + a[3 * HID + j]) + S[i];
}

// ---------------------------------------------------------------------------
// Pooling + FC head
// ---------------------------------------------------------------------------
__global__ void pool_accum(const float* __restrict__ Hf,
                           const int* __restrict__ batch,
                           float* __restrict__ pooled,
                           float* __restrict__ cnt) {
    int n = blockIdx.x * blockDim.x + threadIdx.x;
    if (n >= NNODES) return;
    int g = batch[n];
    atomicAdd(&cnt[g], 1.0f);
    const float* hr = Hf + (size_t)n * HID;
#pragma unroll
    for (int j = 0; j < HID; ++j) atomicAdd(&pooled[(size_t)g * HID + j], hr[j]);
}

__global__ void fc_out(const float* __restrict__ pooled,
                       const float* __restrict__ cnt,
                       const float* __restrict__ fw,
                       const float* __restrict__ fb,
                       float* __restrict__ out) {
    int g = blockIdx.x * blockDim.x + threadIdx.x;
    if (g >= NG) return;
    float inv = 1.0f / fmaxf(cnt[g], 1.0f);
    float acc = fb[0];
#pragma unroll
    for (int j = 0; j < HID; ++j) acc += pooled[(size_t)g * HID + j] * inv * fw[j];
    out[g] = acc;
}

// ---------------------------------------------------------------------------
// Fills
// ---------------------------------------------------------------------------
__global__ void fill_u32_k(unsigned* p, unsigned v, int n) {
    int i = blockIdx.x * blockDim.x + threadIdx.x;
    if (i < n) p[i] = v;
}
__global__ void fill_f32_k(float* p, float v, int n) {
    int i = blockIdx.x * blockDim.x + threadIdx.x;
    if (i < n) p[i] = v;
}

// ---------------------------------------------------------------------------
extern "C" void kernel_launch(void* const* d_in, const int* in_sizes, int n_in,
                              void* d_out, int out_size, void* d_ws, size_t ws_size,
                              hipStream_t stream) {
    (void)in_sizes; (void)n_in; (void)out_size; (void)ws_size;

    const float* x     = (const float*)d_in[0];
    const int*   ei    = (const int*)d_in[1];
    const int*   batch = (const int*)d_in[2];
    const int*   src   = ei;
    const int*   dst   = ei + NEDGES;
    const float* fcw   = (const float*)d_in[27];
    const float* fcb   = (const float*)d_in[28];

    // ---- workspace carve-out (floats) ----
    float* ws = (float*)d_ws;
    size_t off = 0;
    auto alloc = [&](size_t n) { float* p = ws + off; off += n; return p; };
    float*    N0  = alloc((size_t)NNODES * DCAT);   // layer output / next input
    float*    Qb  = alloc((size_t)NNODES * DCAT);   // q, then reused as ATT
    float*    Kb  = alloc((size_t)NNODES * DCAT);
    float*    Vb  = alloc((size_t)NNODES * DCAT);
    float*    Sb  = alloc((size_t)NNODES * DCAT);   // skip
    float*    P   = alloc((size_t)NEDGES * NH);     // logits -> p
    unsigned* Mk  = (unsigned*)alloc((size_t)NNODES * NH);
    float*    Zb  = alloc((size_t)NNODES * NH);
    float*    pooled = alloc((size_t)NG * HID);
    float*    cnt    = alloc(NG);
    _Float16* wh  = (_Float16*)alloc(80000);        // 160000 halves of weight space

    const int Ks[3] = {FIN, DCAT, DCAT};
    const float* Xin = x;
    size_t hoff = 0;
    const int TB = 256;

    for (int li = 0; li < 3; ++li) {
        int K = Ks[li];
        int Mskip = (li < 2) ? DCAT : HID;
        const float* Wq = (const float*)d_in[3 + li * 8 + 0];
        const float* bq = (const float*)d_in[3 + li * 8 + 1];
        const float* Wk = (const float*)d_in[3 + li * 8 + 2];
        const float* bk = (const float*)d_in[3 + li * 8 + 3];
        const float* Wv = (const float*)d_in[3 + li * 8 + 4];
        const float* bv = (const float*)d_in[3 + li * 8 + 5];
        const float* Ws = (const float*)d_in[3 + li * 8 + 6];
        const float* bs = (const float*)d_in[3 + li * 8 + 7];

        _Float16* Wqh = wh + hoff; hoff += (size_t)DCAT * K;
        _Float16* Wkh = wh + hoff; hoff += (size_t)DCAT * K;
        _Float16* Wvh = wh + hoff; hoff += (size_t)DCAT * K;
        _Float16* Wsh = wh + hoff; hoff += (size_t)Mskip * K;

        int ne = K * DCAT;
        convert_w<<<(ne + TB - 1) / TB, TB, 0, stream>>>(Wq, Wqh, K, DCAT);
        convert_w<<<(ne + TB - 1) / TB, TB, 0, stream>>>(Wk, Wkh, K, DCAT);
        convert_w<<<(ne + TB - 1) / TB, TB, 0, stream>>>(Wv, Wvh, K, DCAT);
        int ns = K * Mskip;
        convert_w<<<(ns + TB - 1) / TB, TB, 0, stream>>>(Ws, Wsh, K, Mskip);

        // dense GEMMs (WMMA, weights staged in LDS)
        if (K == FIN) {
            gemm_m128_k64<<<GEMM_GRID, 256, 0, stream>>>(Xin, Wqh, bq, Qb);
            gemm_m128_k64<<<GEMM_GRID, 256, 0, stream>>>(Xin, Wkh, bk, Kb);
            gemm_m128_k64<<<GEMM_GRID, 256, 0, stream>>>(Xin, Wvh, bv, Vb);
            gemm_m128_k64<<<GEMM_GRID, 256, 0, stream>>>(Xin, Wsh, bs, Sb);
        } else {
            gemm_m128_k128<<<GEMM_GRID, 256, 0, stream>>>(Xin, Wqh, bq, Qb);
            gemm_m128_k128<<<GEMM_GRID, 256, 0, stream>>>(Xin, Wkh, bk, Kb);
            gemm_m128_k128<<<GEMM_GRID, 256, 0, stream>>>(Xin, Wvh, bv, Vb);
            if (Mskip == DCAT)
                gemm_m128_k128<<<GEMM_GRID, 256, 0, stream>>>(Xin, Wsh, bs, Sb);
            else
                gemm_m32_k128<<<GEMM_GRID, 256, 0, stream>>>(Xin, Wsh, bs, Sb);
        }

        // segment softmax over incoming edges of dst
        int nnh = NNODES * NH;
        fill_u32_k<<<(nnh + TB - 1) / TB, TB, 0, stream>>>(Mk, 0u, nnh);
        fill_f32_k<<<(nnh + TB - 1) / TB, TB, 0, stream>>>(Zb, 0.0f, nnh);
        edge_logits_max<<<(NEDGES + TB - 1) / TB, TB, 0, stream>>>(Qb, Kb, src, dst, P, Mk);
        int enh = NEDGES * NH;
        edge_softmax_num<<<(enh + TB - 1) / TB, TB, 0, stream>>>(dst, Mk, P, Zb);

        // aggregate: ATT aliases Qb (q dead after pass A)
        int ndc = NNODES * DCAT;
        fill_f32_k<<<(ndc + TB - 1) / TB, TB, 0, stream>>>(Qb, 0.0f, ndc);
        edge_aggregate<<<(enh + TB - 1) / TB, TB, 0, stream>>>(src, dst, P, Zb, Vb, Qb);

        // combine with skip (combine never reads Xin, so N0 overwrite is safe)
        if (li < 2)
            combine_concat_relu<<<(ndc + TB - 1) / TB, TB, 0, stream>>>(Qb, Sb, N0);
        else {
            int nh2 = NNODES * HID;
            combine_mean<<<(nh2 + TB - 1) / TB, TB, 0, stream>>>(Qb, Sb, N0);
        }
        Xin = N0;
    }

    // global mean pool + FC
    int ngh = NG * HID;
    fill_f32_k<<<(ngh + TB - 1) / TB, TB, 0, stream>>>(pooled, 0.0f, ngh);
    fill_f32_k<<<(NG + TB - 1) / TB, TB, 0, stream>>>(cnt, 0.0f, NG);
    pool_accum<<<(NNODES + TB - 1) / TB, TB, 0, stream>>>(N0, batch, pooled, cnt);
    fc_out<<<(NG + TB - 1) / TB, TB, 0, stream>>>(pooled, cnt, fcw, fcb, (float*)d_out);
}
